// MultiHeadAttention_34583076667503
// MI455X (gfx1250) — matrix operand order repaired and checked by reading, not compiled, not run on hardware
//
#include <hip/hip_runtime.h>
#include <stdint.h>

#define DEVI __device__ __forceinline__

typedef __bf16 bf16;
typedef __bf16 bf16x8  __attribute__((ext_vector_type(8)));
typedef __bf16 bf16x16 __attribute__((ext_vector_type(16)));
typedef float  f32x4   __attribute__((ext_vector_type(4)));
typedef float  f32x8   __attribute__((ext_vector_type(8)));
typedef unsigned int u32x4 __attribute__((ext_vector_type(4)));
typedef int          i32x4 __attribute__((ext_vector_type(4)));
typedef int          i32x8 __attribute__((ext_vector_type(8)));

constexpr int Bsz = 2, S = 2048, H = 1024, NH = 16, HD = 64;
constexpr int M = Bsz * S; // 4096 rows

// ---------------------------------------------------------------------------
// WMMA helper (CDNA5 16x16x32 bf16, f32 accumulate)
// ---------------------------------------------------------------------------
DEVI f32x8 wmma_bf16(bf16x16 a, bf16x16 b, f32x8 c) {
  return __builtin_amdgcn_wmma_f32_16x16x32_bf16(false, a, false, b, (short)0, c,
                                                 false, false);
}

// A fragment 16x32 (m = lane%16):
// halves 0-7  -> k = kBase + 8*(lane/16) + 0..7
// halves 8-15 -> k = kBase + 8*(lane/16) + 16..23
DEVI bf16x16 afrag_f32(const float* __restrict__ row, int k, int lane) {
  const int off = k + ((lane >> 4) << 3);
  f32x4 a0 = *(const f32x4*)(row + off);
  f32x4 a1 = *(const f32x4*)(row + off + 4);
  f32x4 b0 = *(const f32x4*)(row + off + 16);
  f32x4 b1 = *(const f32x4*)(row + off + 20);
  bf16x16 r;
#pragma unroll
  for (int i = 0; i < 4; ++i) {
    r[i]      = (bf16)a0[i];
    r[4 + i]  = (bf16)a1[i];
    r[8 + i]  = (bf16)b0[i];
    r[12 + i] = (bf16)b1[i];
  }
  return r;
}

DEVI bf16x16 afrag_bf16(const bf16* __restrict__ row, int k, int lane) {
  const int off = k + ((lane >> 4) << 3);
  bf16x8 lo = *(const bf16x8*)(row + off);
  bf16x8 hi = *(const bf16x8*)(row + off + 16);
  bf16x16 r;
#pragma unroll
  for (int i = 0; i < 8; ++i) { r[i] = lo[i]; r[8 + i] = hi[i]; }
  return r;
}

// B fragment 32x16 (n = lane%16): halves 0-7 -> k = kBase + 8*(lane/16) + 0..7,
// halves 8-15 -> k = kBase + 8*(lane/16) + 16..23
DEVI bf16x16 bfrag_bf16(const bf16* __restrict__ row, int k, int lane) {
  const int off = k + ((lane >> 4) << 3);
  bf16x8 lo = *(const bf16x8*)(row + off);
  bf16x8 hi = *(const bf16x8*)(row + off + 16);
  bf16x16 r;
#pragma unroll
  for (int i = 0; i < 8; ++i) { r[i] = lo[i]; r[8 + i] = hi[i]; }
  return r;
}

// ---------------------------------------------------------------------------
// TDM: 1-D contiguous bf16 tile load, global -> LDS (6-arg builtin on this
// toolchain). Descriptor per CDNA5 ISA ch.8: group0 = {count, lds_addr,
// global_addr, type=2}; group1 = {data_size=2B, tensor_dim0 = tile_dim0 =
// nelem, stride = nelem}; groups 2/3 unused for <=2D tensors.
// ---------------------------------------------------------------------------
DEVI void tdm_load_bf16_1d(uint32_t lds_addr, const bf16* gaddr, uint32_t nelem) {
  const uint64_t ga = (uint64_t)(uintptr_t)gaddr;
  u32x4 g0;
  g0[0] = 1u;                                                  // count=1
  g0[1] = lds_addr;                                            // lds_addr[31:0]
  g0[2] = (uint32_t)ga;                                        // global_addr lo
  g0[3] = (uint32_t)((ga >> 32) & 0x01FFFFFFu) | (2u << 30);   // addr hi, type=2
  i32x8 g1;
  g1[0] = 0x00010000;               // workgroup_mask=0, data_size=1 (2 bytes)
  g1[1] = (int)(nelem << 16);       // tensor_dim0[15:0] in bits [31:16]
  g1[2] = 0x00010000;               // tensor_dim0 hi=0, tensor_dim1=1
  g1[3] = (int)(nelem << 16);       // tensor_dim1 hi=0, tile_dim0 = nelem
  g1[4] = 0;                        // tile_dim1/2 unused
  g1[5] = (int)nelem;               // tensor_dim0_stride lo
  g1[6] = 0;
  g1[7] = 0;
  const i32x4 z4 = {0, 0, 0, 0};
  const i32x8 z8 = {0, 0, 0, 0, 0, 0, 0, 0};
  __builtin_amdgcn_tensor_load_to_lds(g0, g1, z4, z4, z8, 0);
}

// ---------------------------------------------------------------------------
// Kernel 0: f32 -> bf16 weight pre-conversion (one shot per matrix)
// ---------------------------------------------------------------------------
__global__ void __launch_bounds__(256) cvt_bf16_kernel(
    const float* __restrict__ src, bf16* __restrict__ dst) {
  const int i = (blockIdx.x * 256 + threadIdx.x) * 8;
  f32x4 a = *(const f32x4*)(src + i);
  f32x4 b = *(const f32x4*)(src + i + 4);
  bf16x8 o;
#pragma unroll
  for (int j = 0; j < 4; ++j) { o[j] = (bf16)a[j]; o[4 + j] = (bf16)b[j]; }
  *(bf16x8*)(dst + i) = o;
}

// ---------------------------------------------------------------------------
// Kernel 1: X[4096,1024](f32) @ Wb^T(bf16) -> bf16 out in [B, NH, S, HD]
// One wave computes a 16(M) x 64(N) tile. All 4 B fragments are preloaded
// before the WMMA burst so one s_wait covers 8 clause'd loads.
// ---------------------------------------------------------------------------
__global__ void __launch_bounds__(256) qkv_gemm_kernel(
    const float* __restrict__ X, const bf16* __restrict__ Wb,
    bf16* __restrict__ Out) {
  const int lane = threadIdx.x & 31;
  const int wave = threadIdx.x >> 5;
  const int m0 = (blockIdx.y * 8 + wave) * 16;
  const int n0 = blockIdx.x * 64;
  const float* arow = X + (size_t)(m0 + (lane & 15)) * H;
  const bf16* brow0 = Wb + (size_t)(n0 + (lane & 15)) * H;

  f32x8 acc[4] = {};
#pragma unroll 2
  for (int k = 0; k < H; k += 32) {
    __builtin_prefetch(arow + k + 32, 0, 1);
    bf16x16 b[4];
#pragma unroll
    for (int j = 0; j < 4; ++j)
      b[j] = bfrag_bf16(brow0 + (size_t)j * 16 * H, k, lane);
    bf16x16 a = afrag_f32(arow, k, lane);
#pragma unroll
    for (int j = 0; j < 4; ++j) acc[j] = wmma_bf16(a, b[j], acc[j]);
  }

  const int bb = m0 / S;
  const int sbase = (m0 % S) + ((lane >> 4) << 3);
#pragma unroll
  for (int j = 0; j < 4; ++j) {
    const int n = n0 + j * 16 + (lane & 15);
    const int h = n >> 6, d = n & 63;
    bf16* outp = Out + (((size_t)(bb * NH + h) * S + sbase) * HD + d);
#pragma unroll
    for (int r = 0; r < 8; ++r) outp[(size_t)r * HD] = (bf16)acc[j][r];
  }
}

// ---------------------------------------------------------------------------
// Kernel 2: flash attention. Block = one (b,h) x 128 query rows (8 waves x 16).
// K tile staged by TDM (contiguous 4KB), V tile manually transposed into LDS.
// ---------------------------------------------------------------------------
__global__ void __launch_bounds__(256) attn_kernel(
    const bf16* __restrict__ Q, const bf16* __restrict__ K,
    const bf16* __restrict__ V, bf16* __restrict__ ctx) {
  __shared__ alignas(16) bf16 Kt[32 * 64];     // [key][d]
  __shared__ alignas(16) bf16 VtT[64 * 32];    // [d][key]
  __shared__ alignas(16) bf16 Pb[8 * 16 * 32]; // per-wave P staging [q][key]

  const int lane = threadIdx.x & 31;
  const int wave = threadIdx.x >> 5;
  const int grp = lane >> 4;
  const int ln = lane & 15;
  const int bh = blockIdx.x >> 4;   // b*NH + h
  const int qblk = blockIdx.x & 15;
  const int s0 = qblk * 128 + wave * 16;

  const bf16* Qbase = Q + (size_t)bh * S * HD;
  const bf16* Kbase = K + (size_t)bh * S * HD;
  const bf16* Vbase = V + (size_t)bh * S * HD;
  const uint32_t KtLds = (uint32_t)(uintptr_t)&Kt[0];

  const bf16* qrow = Qbase + (size_t)(s0 + ln) * HD;
  const bf16x16 qf0 = afrag_bf16(qrow, 0, lane);
  const bf16x16 qf1 = afrag_bf16(qrow, 32, lane);

  f32x8 oacc[4] = {};
  float rowM[8], rowL[8];
#pragma unroll
  for (int r = 0; r < 8; ++r) { rowM[r] = -1e30f; rowL[r] = 0.f; }

  bf16* Pw = Pb + wave * 512;
  const float scale = 0.125f; // 1/sqrt(64)

  for (int kt = 0; kt < S; kt += 32) {
    __syncthreads(); // previous tile fully consumed before overwrite
    // --- K tile: one TDM op from wave 0 (32x64 bf16 contiguous = 4KB) ---
    if (wave == 0) {
      tdm_load_bf16_1d(KtLds, Kbase + (size_t)kt * HD, 32 * HD);
      __builtin_amdgcn_s_wait_tensorcnt(0);
    }
    // --- V tile: manual transposed staging, 8 bf16 per thread ---
    {
      const int idx = threadIdx.x * 8; // 0..2047
      const int krow = idx >> 6;
      const int kcol = idx & 63;
      const bf16* gv = Vbase + (size_t)(kt + krow) * HD + kcol;
      bf16x8 v0 = *(const bf16x8*)gv;
#pragma unroll
      for (int i = 0; i < 8; ++i) VtT[(kcol + i) * 32 + krow] = v0[i];
    }
    __syncthreads();

    // logits for 32 keys: two 16x16 tiles, K-dim = HD = 64
    f32x8 sa = {}, sb = {};
    {
      const bf16* k0 = Kt + (size_t)ln * 64;
      const bf16* k1 = Kt + (size_t)(16 + ln) * 64;
      bf16x16 kb00 = bfrag_bf16(k0, 0, lane);
      bf16x16 kb01 = bfrag_bf16(k0, 32, lane);
      bf16x16 kb10 = bfrag_bf16(k1, 0, lane);
      bf16x16 kb11 = bfrag_bf16(k1, 32, lane);
      sa = wmma_bf16(qf0, kb00, sa);
      sa = wmma_bf16(qf1, kb01, sa);
      sb = wmma_bf16(qf0, kb10, sb);
      sb = wmma_bf16(qf1, kb11, sb);
    }

    // online softmax (row-wise across the 16 lanes of each half-group)
#pragma unroll
    for (int r = 0; r < 8; ++r) {
      float v = fmaxf(sa[r], sb[r]);
#pragma unroll
      for (int off = 1; off < 16; off <<= 1) v = fmaxf(v, __shfl_xor(v, off));
      v *= scale;
      const float nm = fmaxf(rowM[r], v);
      const float corr = __expf(rowM[r] - nm);
      rowM[r] = nm;
      const float p0 = __expf(sa[r] * scale - nm);
      const float p1 = __expf(sb[r] * scale - nm);
      float ps = p0 + p1;
#pragma unroll
      for (int off = 1; off < 16; off <<= 1) ps += __shfl_xor(ps, off);
      rowL[r] = rowL[r] * corr + ps;
#pragma unroll
      for (int j = 0; j < 4; ++j) oacc[j][r] *= corr;
      const int mrow = r + (grp << 3);
      Pw[mrow * 32 + ln] = (bf16)p0;
      Pw[mrow * 32 + 16 + ln] = (bf16)p1;
    }

    // O += P @ V (A = P via LDS relayout, B = V^T rows from LDS)
    const bf16x16 pf = afrag_bf16(Pw + (size_t)ln * 32, 0, lane);
    bf16x16 vb[4];
#pragma unroll
    for (int j = 0; j < 4; ++j)
      vb[j] = bfrag_bf16(VtT + (size_t)(j * 16 + ln) * 32, 0, lane);
#pragma unroll
    for (int j = 0; j < 4; ++j) oacc[j] = wmma_bf16(pf, vb[j], oacc[j]);
  }

  // epilogue: O / l, store ctx as bf16 in [B, S, NH*HD] row-major
  const int bb = bh >> 4;
  const int h = bh & 15;
#pragma unroll
  for (int j = 0; j < 4; ++j) {
#pragma unroll
    for (int r = 0; r < 8; ++r) {
      const int s = s0 + r + (grp << 3);
      const int col = h * HD + j * 16 + ln;
      ctx[(size_t)(bb * S + s) * H + col] = (bf16)(oacc[j][r] / rowL[r]);
    }
  }
}

// ---------------------------------------------------------------------------
// Kernel 3: resid = query + ctx @ Wo^T + bo (fp32 out). Same tiling as k1.
// ---------------------------------------------------------------------------
__global__ void __launch_bounds__(256) proj_resid_kernel(
    const bf16* __restrict__ ctx, const bf16* __restrict__ Wob,
    const float* __restrict__ bo, const float* __restrict__ query,
    float* __restrict__ resid) {
  const int lane = threadIdx.x & 31;
  const int wave = threadIdx.x >> 5;
  const int m0 = (blockIdx.y * 8 + wave) * 16;
  const int n0 = blockIdx.x * 64;
  const bf16* arow  = ctx + (size_t)(m0 + (lane & 15)) * H;
  const bf16* brow0 = Wob + (size_t)(n0 + (lane & 15)) * H;

  f32x8 acc[4] = {};
#pragma unroll 2
  for (int k = 0; k < H; k += 32) {
    bf16x16 b[4];
#pragma unroll
    for (int j = 0; j < 4; ++j)
      b[j] = bfrag_bf16(brow0 + (size_t)j * 16 * H, k, lane);
    bf16x16 a = afrag_bf16(arow, k, lane);
#pragma unroll
    for (int j = 0; j < 4; ++j) acc[j] = wmma_bf16(a, b[j], acc[j]);
  }

  const int grp = lane >> 4;
#pragma unroll
  for (int j = 0; j < 4; ++j) {
    const int n = n0 + j * 16 + (lane & 15);
    const float bias = bo[n];
#pragma unroll
    for (int r = 0; r < 8; ++r) {
      const int m = m0 + r + (grp << 3);
      resid[(size_t)m * H + n] = acc[j][r] + bias + query[(size_t)m * H + n];
    }
  }
}

// ---------------------------------------------------------------------------
// Kernel 4: layernorm over H=1024 per row. One block (256 thr) per row.
// ---------------------------------------------------------------------------
__global__ void __launch_bounds__(256) layernorm_kernel(
    const float* __restrict__ resid, const float* __restrict__ gamma,
    const float* __restrict__ beta, float* __restrict__ out) {
  const int row = blockIdx.x;
  const int t = threadIdx.x;
  const float* rp = resid + (size_t)row * H;
  f32x4 x = *(const f32x4*)(rp + t * 4);
  float s = x[0] + x[1] + x[2] + x[3];
  float q = x[0] * x[0] + x[1] * x[1] + x[2] * x[2] + x[3] * x[3];
#pragma unroll
  for (int off = 16; off >= 1; off >>= 1) {
    s += __shfl_xor(s, off);
    q += __shfl_xor(q, off);
  }
  __shared__ float reds[8], redq[8];
  const int wv = t >> 5;
  if ((t & 31) == 0) { reds[wv] = s; redq[wv] = q; }
  __syncthreads();
  float ts = 0.f, tq = 0.f;
#pragma unroll
  for (int i = 0; i < 8; ++i) { ts += reds[i]; tq += redq[i]; }
  const float mu = ts * (1.0f / H);
  const float var = tq * (1.0f / H) - mu * mu;
  const float rinv = rsqrtf(var + 1e-5f);
  f32x4 o;
#pragma unroll
  for (int i = 0; i < 4; ++i) {
    const int c = t * 4 + i;
    o[i] = (x[i] - mu) * rinv * gamma[c] + beta[c];
  }
  *(f32x4*)(out + (size_t)row * H + t * 4) = o;
}

// ---------------------------------------------------------------------------
extern "C" void kernel_launch(void* const* d_in, const int* in_sizes, int n_in,
                              void* d_out, int out_size, void* d_ws,
                              size_t ws_size, hipStream_t stream) {
  const float* query = (const float*)d_in[0];
  const float* key   = (const float*)d_in[1];
  const float* value = (const float*)d_in[2];
  const float* Wq    = (const float*)d_in[3];
  const float* Wk    = (const float*)d_in[4];
  const float* Wv    = (const float*)d_in[5];
  const float* Wo    = (const float*)d_in[6];
  const float* bo    = (const float*)d_in[7];
  const float* ln_g  = (const float*)d_in[8];
  const float* ln_b  = (const float*)d_in[9];

  char* ws = (char*)d_ws;
  bf16* Qh     = (bf16*)(ws);                        // 8 MB [B,NH,S,HD] bf16
  bf16* Kh     = (bf16*)(ws + ((size_t)8 << 20));    // 8 MB
  bf16* Vh     = (bf16*)(ws + ((size_t)16 << 20));   // 8 MB
  bf16* ctx    = (bf16*)(ws + ((size_t)24 << 20));   // 8 MB [B*S, H] bf16
  float* resid = (float*)(ws + ((size_t)32 << 20));  // 16 MB [B*S, H] f32
  bf16* Wqb    = (bf16*)(ws + ((size_t)48 << 20));   // 2 MB each
  bf16* Wkb    = (bf16*)(ws + ((size_t)50 << 20));
  bf16* Wvb    = (bf16*)(ws + ((size_t)52 << 20));
  bf16* Wob    = (bf16*)(ws + ((size_t)54 << 20));
  float* out = (float*)d_out;

  // one-shot weight conversion: H*H / (256*8) = 512 blocks each
  dim3 cgrid(H * H / (256 * 8));
  cvt_bf16_kernel<<<cgrid, 256, 0, stream>>>(Wq, Wqb);
  cvt_bf16_kernel<<<cgrid, 256, 0, stream>>>(Wk, Wkb);
  cvt_bf16_kernel<<<cgrid, 256, 0, stream>>>(Wv, Wvb);
  cvt_bf16_kernel<<<cgrid, 256, 0, stream>>>(Wo, Wob);

  dim3 gblk(H / 64, M / 128); // (16, 32), 256 threads = 8 waves
  qkv_gemm_kernel<<<gblk, 256, 0, stream>>>(query, Wqb, Qh);
  qkv_gemm_kernel<<<gblk, 256, 0, stream>>>(key, Wkb, Kh);
  qkv_gemm_kernel<<<gblk, 256, 0, stream>>>(value, Wvb, Vh);

  attn_kernel<<<dim3(Bsz * NH * (S / 128)), 256, 0, stream>>>(Qh, Kh, Vh, ctx);

  proj_resid_kernel<<<gblk, 256, 0, stream>>>(ctx, Wob, bo, query, resid);
  layernorm_kernel<<<dim3(M), 256, 0, stream>>>(resid, ln_g, ln_b, out);
}
